// Attention_75857712382074
// MI455X (gfx1250) — matrix operand order repaired and checked by reading, not compiled, not run on hardware
//
#include <hip/hip_runtime.h>
#include <hip/hip_bf16.h>
#include <cstdint>

typedef __attribute__((ext_vector_type(16))) __bf16 v16bf;
typedef __attribute__((ext_vector_type(8)))  __bf16 bf16x8;
typedef __attribute__((ext_vector_type(4)))  __bf16 bf16x4;
typedef __attribute__((ext_vector_type(8)))  float  v8f;
typedef __attribute__((ext_vector_type(4)))  float  f32x4;

static __device__ __forceinline__ v16bf cat8(bf16x8 lo, bf16x8 hi) {
    return __builtin_shufflevector(lo, hi, 0,1,2,3,4,5,6,7,8,9,10,11,12,13,14,15);
}
static __device__ __forceinline__ v8f zero8() {
    v8f z;
#pragma unroll
    for (int i = 0; i < 8; ++i) z[i] = 0.f;
    return z;
}

// CDNA5 async global->LDS copy (ASYNCcnt path). GV mode: mem_addr = VADDR[63:0].
// VDST carries the per-lane LDS byte offset (low 32 bits of the generic pointer:
// flat LDS addresses are aperture[63:32] | offset[31:0], ISA 10.2).
static __device__ __forceinline__ void async_copy_b128(void* lds_dst, const void* gsrc) {
    unsigned     ldsoff = (unsigned)(uintptr_t)lds_dst;
    unsigned long long ga = (unsigned long long)(uintptr_t)gsrc;
    asm volatile("global_load_async_to_lds_b128 %0, %1, off"
                 :: "v"(ldsoff), "v"(ga) : "memory");
}
static __device__ __forceinline__ void wait_async0() {
#if __has_builtin(__builtin_amdgcn_s_wait_asynccnt)
    __builtin_amdgcn_s_wait_asynccnt(0);
#else
    asm volatile("s_wait_asynccnt 0" ::: "memory");
#endif
}

// ---------------------------------------------------------------------------
// Generic tiled GEMM: C[M,N] = A[M,K] @ B[K,N] (+ bias), bf16 WMMA, f32 accum.
// Block: 256 threads = 8 waves in a 4(M) x 2(N) grid; wave tile 32x64.
// Requires M%128==0, N%128==0, K%32==0 (true for all our shapes).
// ---------------------------------------------------------------------------
template <typename AT, typename OT, bool BIAS>
__global__ __launch_bounds__(256) void gemm_wmma(
    const AT* __restrict__ A, const float* __restrict__ B,
    OT* __restrict__ C, const float* __restrict__ bias,
    int M, int N, int K) {
    constexpr int BM = 128, BN = 128, BK = 32, LDW = 40;  // padded LDS row stride
    __shared__ __bf16 As[BM][LDW];
    __shared__ __bf16 Bs[BN][LDW];  // stored transposed: Bs[n][k]

    const int t    = threadIdx.x;
    const int lane = t & 31;
    const int wave = t >> 5;      // 0..7
    const int wm   = wave >> 1;   // 0..3  (M direction)
    const int wn   = wave & 1;    // 0..1  (N direction)
    const int bm   = blockIdx.y * BM;
    const int bn   = blockIdx.x * BN;
    const int half = lane >> 4;   // 0/1
    const int l16  = lane & 15;
    const int kbA  = half * 8;    // A-fragment K base (chunks at kbA, kbA+16)
    const int kbB  = half * 8;    // B-fragment K base (chunks at kbB, kbB+16)

    v8f acc[2][4];
#pragma unroll
    for (int i = 0; i < 2; ++i)
#pragma unroll
        for (int j = 0; j < 4; ++j) acc[i][j] = zero8();

    for (int k0 = 0; k0 < K; k0 += BK) {
        __syncthreads();
        // ---- stage A tile (128 x 32) into LDS as bf16 ----
        if constexpr (__is_same(AT, float)) {
#pragma unroll
            for (int i = 0; i < 4; ++i) {
                int f  = t + 256 * i;          // 1024 float4 groups
                int r  = f >> 3;               // 8 float4 per 32-wide row
                int c4 = (f & 7) * 4;
                f32x4 v = *reinterpret_cast<const f32x4*>(&A[(size_t)(bm + r) * K + k0 + c4]);
                bf16x4 p;
                p[0] = (__bf16)v.x; p[1] = (__bf16)v.y;
                p[2] = (__bf16)v.z; p[3] = (__bf16)v.w;
                *reinterpret_cast<bf16x4*>(&As[r][c4]) = p;   // one b64 LDS store
            }
        } else {
            // bf16 activations: DMA straight into LDS, no VGPR round-trip.
#pragma unroll
            for (int i = 0; i < 2; ++i) {
                int f  = t + 256 * i;          // 512 bf16x8 groups
                int r  = f >> 2;               // 4 groups per row
                int c8 = (f & 3) * 8;
                async_copy_b128(&As[r][c8], &A[(size_t)(bm + r) * K + k0 + c8]);
            }
        }
        // ---- stage B tile (32 x 128) transposed: column loads, packed b64 stores ----
#pragma unroll
        for (int i = 0; i < 4; ++i) {
            int f  = t + 256 * i;              // 1024 (n, k4) cells
            int n  = f & 127;                  // coalesced across threads
            int k4 = (f >> 7) * 4;             // 0,4,...,28
            bf16x4 p;
#pragma unroll
            for (int j = 0; j < 4; ++j)
                p[j] = (__bf16)B[(size_t)(k0 + k4 + j) * N + bn + n];
            *reinterpret_cast<bf16x4*>(&Bs[n][k4]) = p;       // one b64 LDS store
        }
        if constexpr (!__is_same(AT, float)) wait_async0();
        __syncthreads();

        // ---- build fragments and issue WMMAs ----
        v16bf af[2], bfr[4];
#pragma unroll
        for (int sm = 0; sm < 2; ++sm) {
            int r = wm * 32 + sm * 16 + l16;
            bf16x8 lo = *reinterpret_cast<const bf16x8*>(&As[r][kbA]);
            bf16x8 hi = *reinterpret_cast<const bf16x8*>(&As[r][kbA + 16]);
            af[sm] = cat8(lo, hi);
        }
#pragma unroll
        for (int sn = 0; sn < 4; ++sn) {
            int c = wn * 64 + sn * 16 + l16;
            bf16x8 lo = *reinterpret_cast<const bf16x8*>(&Bs[c][kbB]);
            bf16x8 hi = *reinterpret_cast<const bf16x8*>(&Bs[c][kbB + 16]);
            bfr[sn] = cat8(lo, hi);
        }
#pragma unroll
        for (int sm = 0; sm < 2; ++sm)
#pragma unroll
            for (int sn = 0; sn < 4; ++sn)
                acc[sm][sn] = __builtin_amdgcn_wmma_f32_16x16x32_bf16(
                    false, af[sm], false, bfr[sn], (short)0, acc[sm][sn], false, false);
    }

    // ---- epilogue: C/D layout row = v + 8*half, col = l16 ----
#pragma unroll
    for (int sm = 0; sm < 2; ++sm) {
#pragma unroll
        for (int sn = 0; sn < 4; ++sn) {
            int col = bn + wn * 64 + sn * 16 + l16;
            float bv = BIAS ? bias[col] : 0.f;
#pragma unroll
            for (int v = 0; v < 8; ++v) {
                int row = bm + wm * 32 + sm * 16 + v + half * 8;
                C[(size_t)row * N + col] = (OT)(acc[sm][sn][v] + bv);
            }
        }
    }
}

// ---------------------------------------------------------------------------
// Flash attention over bf16 Q/K/V projections. One block per (batch, head).
// 8 waves x 16 query rows; stream M=2048 keys in 64-key chunks.
// K chunk staged once per block via async global->LDS (was read 8x before);
// V chunk staged transposed via VALU (overlaps the async K DMA).
// ---------------------------------------------------------------------------
__global__ __launch_bounds__(256) void attn_wmma(
    const __bf16* __restrict__ Qp, const __bf16* __restrict__ Kp,
    const __bf16* __restrict__ Vp, __bf16* __restrict__ ctx) {
    constexpr int N = 128, M = 2048, D = 64, INNER = 1024, CH = 64;
    const int bh   = blockIdx.x;
    const int b    = bh >> 4;
    const int h    = bh & 15;
    const int t    = threadIdx.x;
    const int lane = t & 31;
    const int wave = t >> 5;
    const int half = lane >> 4;
    const int l16  = lane & 15;
    const int kbA  = half * 8;
    const int kbB  = half * 8;
    const float scale = 0.125f;  // 1/sqrt(64)

    __shared__ float  sS[8][16][64];
    __shared__ __bf16 sP[8][16][64];
    __shared__ float  sAlpha[8][16];
    __shared__ float  sLinv[8][16];
    __shared__ __bf16 sVt[64][80];  // V chunk transposed: [d][m], padded row
    __shared__ __bf16 sK[64][80];   // K chunk row-major:  [m][d], padded row

    const size_t qbase  = ((size_t)b * N) * INNER + h * D;
    const size_t kvbase = ((size_t)b * M) * INNER + h * D;

    // Persistent Q fragments for this wave's 16 rows (A layout, two K=32 slices).
    v16bf qf[2];
#pragma unroll
    for (int s = 0; s < 2; ++s) {
        const __bf16* qrow = Qp + qbase + (size_t)(wave * 16 + l16) * INNER + s * 32;
        bf16x8 lo = *reinterpret_cast<const bf16x8*>(qrow + kbA);
        bf16x8 hi = *reinterpret_cast<const bf16x8*>(qrow + kbA + 16);
        qf[s] = cat8(lo, hi);
    }

    v8f oacc[4];
#pragma unroll
    for (int j = 0; j < 4; ++j) oacc[j] = zero8();
    float mrun = -1e30f;  // running row max (row = l16, both halves track it)
    float lrun = 0.f;     // running row sum

    for (int m0 = 0; m0 < M; m0 += CH) {
        // ---- async DMA of K chunk (64 keys x 64 dims) into LDS, row-major ----
#pragma unroll
        for (int i = 0; i < 2; ++i) {
            int f  = t + 256 * i;  // 512 b128 groups
            int m  = f >> 3;
            int d8 = (f & 7) * 8;
            async_copy_b128(&sK[m][d8], &Kp[kvbase + (size_t)(m0 + m) * INNER + d8]);
        }
        // ---- cooperative transposed stage of V chunk (overlaps the DMA) ----
        // K/V bytes are consumed exactly once by exactly one block -> non-temporal.
#pragma unroll
        for (int i = 0; i < 2; ++i) {
            int f  = t + 256 * i;  // 512 bf16x8 groups
            int m  = f >> 3;       // key in chunk
            int d8 = (f & 7) * 8;
            bf16x8 v = __builtin_nontemporal_load(reinterpret_cast<const bf16x8*>(
                &Vp[kvbase + (size_t)(m0 + m) * INNER + d8]));
#pragma unroll
            for (int e = 0; e < 8; ++e) sVt[d8 + e][m] = v[e];
        }
        if (m0 + CH < M)  // hint next V chunk toward the caches
            __builtin_prefetch(&Vp[kvbase + (size_t)(m0 + CH + lane) * INNER], 0, 0);
        wait_async0();
        __syncthreads();

        // ---- S = scale * Q K^T for 16 rows x 64 keys (WMMA) ----
#pragma unroll
        for (int j = 0; j < 4; ++j) {
            v8f sacc = zero8();
#pragma unroll
            for (int s = 0; s < 2; ++s) {
                // B = K^T: lane col = key, K-dim contiguous along d in sK's row.
                const __bf16* krow = &sK[j * 16 + l16][s * 32];
                bf16x8 lo = *reinterpret_cast<const bf16x8*>(krow + kbB);
                bf16x8 hi = *reinterpret_cast<const bf16x8*>(krow + kbB + 16);
                v16bf kf = cat8(lo, hi);
                sacc = __builtin_amdgcn_wmma_f32_16x16x32_bf16(
                    false, qf[s], false, kf, (short)0, sacc, false, false);
            }
#pragma unroll
            for (int v = 0; v < 8; ++v)
                sS[wave][v + half * 8][j * 16 + l16] = sacc[v] * scale;
        }
        __syncthreads();

        // ---- online softmax: lane handles row l16, cols half*32..+31 ----
        {
            float vals[32];
            float mloc = -1e30f;
#pragma unroll
            for (int c = 0; c < 32; ++c) {
                vals[c] = sS[wave][l16][half * 32 + c];
                mloc = fmaxf(mloc, vals[c]);
            }
            mloc = fmaxf(mloc, __shfl_xor(mloc, 16));
            float mnew = fmaxf(mrun, mloc);
            float lsum = 0.f;
#pragma unroll
            for (int c = 0; c < 32; ++c) {
                float p = __expf(vals[c] - mnew);
                lsum += p;
                sP[wave][l16][half * 32 + c] = (__bf16)p;
            }
            lsum += __shfl_xor(lsum, 16);
            float alpha = __expf(mrun - mnew);
            lrun = lrun * alpha + lsum;
            mrun = mnew;
            if (half == 0) sAlpha[wave][l16] = alpha;
        }
        __syncthreads();

        // ---- rescale O accumulators by per-row alpha ----
#pragma unroll
        for (int v = 0; v < 8; ++v) {
            float a = sAlpha[wave][v + half * 8];
#pragma unroll
            for (int j = 0; j < 4; ++j) oacc[j][v] *= a;
        }

        // ---- O += P @ Vchunk (WMMA) ----
#pragma unroll
        for (int s = 0; s < 2; ++s) {
            bf16x8 plo = *reinterpret_cast<const bf16x8*>(&sP[wave][l16][s * 32 + kbA]);
            bf16x8 phi = *reinterpret_cast<const bf16x8*>(&sP[wave][l16][s * 32 + kbA + 16]);
            v16bf pf = cat8(plo, phi);
#pragma unroll
            for (int j = 0; j < 4; ++j) {
                bf16x8 vlo = *reinterpret_cast<const bf16x8*>(&sVt[j * 16 + l16][s * 32 + kbB]);
                bf16x8 vhi = *reinterpret_cast<const bf16x8*>(&sVt[j * 16 + l16][s * 32 + kbB + 16]);
                v16bf vf = cat8(vlo, vhi);
                oacc[j] = __builtin_amdgcn_wmma_f32_16x16x32_bf16(
                    false, pf, false, vf, (short)0, oacc[j], false, false);
            }
        }
        __syncthreads();  // before next chunk overwrites sVt / sK / sS / sP
    }

    // ---- finalize: O /= l, write ctx[b][n][h*64+d] as bf16 ----
    if (half == 0) sLinv[wave][l16] = 1.f / lrun;
    __syncthreads();
#pragma unroll
    for (int v = 0; v < 8; ++v) {
        int row = wave * 16 + v + half * 8;
        float linv = sLinv[wave][v + half * 8];
#pragma unroll
        for (int j = 0; j < 4; ++j) {
            int d = j * 16 + l16;
            ctx[qbase + (size_t)row * INNER + d] = (__bf16)(oacc[j][v] * linv);
        }
    }
}

// ---------------------------------------------------------------------------
// Orchestration
// ---------------------------------------------------------------------------
extern "C" void kernel_launch(void* const* d_in, const int* in_sizes, int n_in,
                              void* d_out, int out_size, void* d_ws, size_t ws_size,
                              hipStream_t stream) {
    const float* q  = (const float*)d_in[0];
    const float* kv = (const float*)d_in[1];
    const float* Wq = (const float*)d_in[2];
    const float* Wk = (const float*)d_in[3];
    const float* Wv = (const float*)d_in[4];
    const float* Wo = (const float*)d_in[5];
    const float* bo = (const float*)d_in[6];
    float* out = (float*)d_out;

    constexpr int B = 32, N = 128, M = 2048, INNER = 1024, QD = 1024, KVD = 512;

    // bf16 workspace: Qp(8MB) | Kp(128MB) | Vp(128MB) | ctx(8MB)
    __bf16* Qp = (__bf16*)d_ws;
    __bf16* Kp = Qp + (size_t)B * N * INNER;
    __bf16* Vp = Kp + (size_t)B * M * INNER;
    __bf16* Cx = Vp + (size_t)B * M * INNER;

    dim3 blk(256);
    // Projections (fp32 in -> bf16 out)
    gemm_wmma<float, __bf16, false>
        <<<dim3(INNER / 128, (B * N) / 128), blk, 0, stream>>>(q, Wq, Qp, nullptr, B * N, INNER, QD);
    gemm_wmma<float, __bf16, false>
        <<<dim3(INNER / 128, (B * M) / 128), blk, 0, stream>>>(kv, Wk, Kp, nullptr, B * M, INNER, KVD);
    gemm_wmma<float, __bf16, false>
        <<<dim3(INNER / 128, (B * M) / 128), blk, 0, stream>>>(kv, Wv, Vp, nullptr, B * M, INNER, KVD);
    // Attention: one block per (batch, head)
    attn_wmma<<<dim3(B * 16), blk, 0, stream>>>(Qp, Kp, Vp, Cx);
    // Output projection (bf16 in -> fp32 out, + bias)
    gemm_wmma<__bf16, float, true>
        <<<dim3(QD / 128, (B * N) / 128), blk, 0, stream>>>(Cx, Wo, out, bo, B * N, QD, INNER);
}
